// RWA_31138512896781
// MI455X (gfx1250) — compile-verified, matrix operand order checked by reading, not run on hardware
//
#include <hip/hip_runtime.h>
#include <hip/hip_bf16.h>

// Problem dims (match reference)
#define B_    64
#define T_    512
#define F_    256
#define C_    1024
#define K_    256
#define FC_   (F_ + C_)      // 1280
#define M_    (B_ * T_)      // 32768 rows of X
#define NCAT_ 4096           // 4 fused F->C matrices: [Wu | Wg_x | Wa_x | Wd]
#define NWG_  16             // persistent workgroups in the recurrent phase

typedef __bf16 bf16_t;
typedef __attribute__((ext_vector_type(16))) __bf16 v16bf;
typedef __attribute__((ext_vector_type(8)))  __bf16 v8bf;
typedef __attribute__((ext_vector_type(8)))  float  v8f;

// ---------------------------------------------------------------------------
// Fragment loader for 16-bit A/B matrices (16x32 MxK per ISA layout):
//   lanes 0-15:  row = lane,    VGPR0-3 hold K=0..7,  VGPR4-7 hold K=16..23
//   lanes 16-31: row = lane-16, VGPR0-3 hold K=8..15, VGPR4-7 hold K=24..31
// Works identically for global and LDS pointers (ds_load_b128 when LDS).
// ---------------------------------------------------------------------------
__device__ __forceinline__ v16bf load_frag_b16(const bf16_t* __restrict__ base,
                                               int row, int ld, int kbase, int lane) {
  const bf16_t* p = base + (size_t)row * ld + kbase;
  const int hi = (lane >= 16) ? 8 : 0;
  v8bf lo = *(const v8bf*)(p + hi);        // K = hi .. hi+7
  v8bf hh = *(const v8bf*)(p + 16 + hi);   // K = 16+hi .. 23+hi
  v16bf r;
#pragma unroll
  for (int i = 0; i < 8; ++i) { r[i] = lo[i]; r[8 + i] = hh[i]; }
  return r;
}

__device__ __forceinline__ v8f wmma_bf16(v16bf a, v16bf b, v8f c) {
  return __builtin_amdgcn_wmma_f32_16x16x32_bf16(false, a, false, b, (short)0, c,
                                                 false, false);
}

// CDNA5 async copy: 16B global -> LDS per lane, tracked by ASYNCcnt.
// LDS byte address = low 32 bits of the generic pointer (aperture mapping).
__device__ __forceinline__ void async_g2l_b128(void* lds_ptr, const void* gptr) {
  uint32_t laddr = (uint32_t)(uintptr_t)lds_ptr;
  asm volatile("global_load_async_to_lds_b128 %0, %1, off"
               :: "v"(laddr), "v"(gptr) : "memory");
}
__device__ __forceinline__ void wait_async0() {
  asm volatile("s_wait_asynccnt 0x0" ::: "memory");
}

// ---------------------------------------------------------------------------
// Packing kernels (per call; ws is not persistent across calls).
// ---------------------------------------------------------------------------
__global__ void pack_x_kernel(const float* __restrict__ x, bf16_t* __restrict__ Xbf) {
  int tid = blockIdx.x * blockDim.x + threadIdx.x;
  int nth = gridDim.x * blockDim.x;
  for (size_t i = tid; i < (size_t)M_ * F_; i += nth) Xbf[i] = (bf16_t)x[i];
}

__global__ void pack_w_kernel(const float* __restrict__ Wu, const float* __restrict__ Wg,
                              const float* __restrict__ Wa, const float* __restrict__ Wd,
                              const float* __restrict__ Wo,
                              bf16_t* __restrict__ Wcat, bf16_t* __restrict__ Wgh,
                              bf16_t* __restrict__ Wah,  bf16_t* __restrict__ Wob) {
  int tid = blockIdx.x * blockDim.x + threadIdx.x;
  int nth = gridDim.x * blockDim.x;
  for (int i = tid; i < NCAT_ * F_; i += nth) {       // Wcat[row=which*C+c][k<F]
    int row = i / F_, k = i - row * F_;
    int which = row >> 10, c = row & (C_ - 1);
    float v = (which == 0) ? Wu[(size_t)c * F_ + k]
            : (which == 1) ? Wg[(size_t)c * FC_ + k]
            : (which == 2) ? Wa[(size_t)c * FC_ + k]
                           : Wd[(size_t)c * F_ + k];
    Wcat[i] = (bf16_t)v;
  }
  for (int i = tid; i < C_ * C_; i += nth) {          // h-recurrence weights
    int c = i >> 10, k = i & (C_ - 1);
    Wgh[i] = (bf16_t)Wg[(size_t)c * FC_ + F_ + k];
    Wah[i] = (bf16_t)Wa[(size_t)c * FC_ + F_ + k];
  }
  for (int i = tid; i < K_ * C_; i += nth) Wob[i] = (bf16_t)Wo[i];
}

// h0 = h + tanh(s); copy recurrent state into ws; emit `s` output segment.
__global__ void init_kernel(const float* __restrict__ s, const float* __restrict__ n0,
                            const float* __restrict__ d0, const float* __restrict__ h0,
                            const float* __restrict__ am0,
                            bf16_t* __restrict__ hbuf, float* __restrict__ nst,
                            float* __restrict__ dst, float* __restrict__ hst,
                            float* __restrict__ amst, float* __restrict__ s_out) {
  int i = blockIdx.x * blockDim.x + threadIdx.x;
  if (i < C_) s_out[i] = s[i];
  if (i < B_ * C_) {
    int c = i & (C_ - 1);
    float h = h0[i] + tanhf(s[c]);
    nst[i] = n0[i]; dst[i] = d0[i]; amst[i] = am0[i]; hst[i] = h;
    hbuf[i] = (bf16_t)h;
  }
}

// ---------------------------------------------------------------------------
// Phase 1: Y[m][ncat] = X @ Wcat^T, fused bias/activation.
// Block = 8 waves, tile 256(M) x 64(N). B strip (64x256 bf16 = 32KB) is
// async-staged to LDS ONCE (K=256 is the whole reduction), A is held in
// registers for the full K. Inner loop: ds_load + wmma only.
// ---------------------------------------------------------------------------
__global__ __launch_bounds__(256) void xgemm_kernel(
    const bf16_t* __restrict__ Xbf, const bf16_t* __restrict__ Wcat,
    const float* __restrict__ bu, const float* __restrict__ bg,
    float* __restrict__ Y) {
  __shared__ __align__(16) bf16_t Bs[64 * F_];        // 32 KB

  const int lane  = threadIdx.x & 31;
  const int wave  = threadIdx.x >> 5;
  const int mbase = blockIdx.x * 256 + wave * 32;     // wave owns 32 M rows
  const int nbase = blockIdx.y * 64;

  // ---- async-stage B strip: 64 rows x 512B; 4 threads/row, 128B each ------
  {
    int row  = threadIdx.x >> 2;
    int quad = (threadIdx.x & 3) * 128;
    const char* g = (const char*)(Wcat + (size_t)(nbase + row) * F_) + quad;
    char* l = (char*)Bs + row * 512 + quad;
#pragma unroll
    for (int i = 0; i < 8; ++i) async_g2l_b128(l + i * 16, g + i * 16);
    wait_async0();
  }
  __syncthreads();

  // ---- A resident in registers for full K (two 16-row strips) -------------
  const int ar0 = mbase + (lane & 15);
  v16bf a0[8], a1[8];
#pragma unroll
  for (int ks = 0; ks < 8; ++ks) {
    a0[ks] = load_frag_b16(Xbf, ar0,      F_, ks * 32, lane);
    a1[ks] = load_frag_b16(Xbf, ar0 + 16, F_, ks * 32, lane);
  }

  v8f acc0[4] = {v8f{}, v8f{}, v8f{}, v8f{}};
  v8f acc1[4] = {v8f{}, v8f{}, v8f{}, v8f{}};
#pragma unroll
  for (int j = 0; j < 4; ++j) {
#pragma unroll
    for (int ks = 0; ks < 8; ++ks) {
      v16bf b = load_frag_b16(Bs, j * 16 + (lane & 15), F_, ks * 32, lane);
      acc0[j] = wmma_bf16(a0[ks], b, acc0[j]);
      acc1[j] = wmma_bf16(a1[ks], b, acc1[j]);
    }
  }

  // ---- fused epilogue, nontemporal stores (Y is a 512MB write-once stream)
  const int m0 = mbase + ((lane < 16) ? 0 : 8);
#pragma unroll
  for (int j = 0; j < 4; ++j) {
    int ncat = nbase + j * 16 + (lane & 15);
    int which = ncat >> 10, c = ncat & (C_ - 1);
    float badd = (which == 0) ? bu[c] : (which == 1) ? bg[c] : 0.0f;
#pragma unroll
    for (int r = 0; r < 8; ++r) {
      float v0 = acc0[j][r] + badd;
      float v1 = acc1[j][r] + badd;
      if (which == 3) { v0 = __expf(tanhf(v0)); v1 = __expf(tanhf(v1)); }
      __builtin_nontemporal_store(v0, &Y[(size_t)(m0 + r) * NCAT_ + ncat]);
      __builtin_nontemporal_store(v1, &Y[(size_t)(m0 + 16 + r) * NCAT_ + ncat]);
    }
  }
}

// ---------------------------------------------------------------------------
// Phase 2: persistent recurrent kernel, NWG_=16 workgroups (cluster-sized,
// single-SE story: the cluster barrier is the intended hardware sync; the
// atomic rel/acq barrier is the portable fallback). Per step, WG owns a
// 64-channel slice. GEMM k-loops are software-pipelined (load k+32 while
// computing k) so s_wait covers a full L2-latency iteration.
// ---------------------------------------------------------------------------
__global__ __launch_bounds__(256) void recurrent_kernel(
    const float* __restrict__ Y, const bf16_t* __restrict__ Wgh,
    const bf16_t* __restrict__ Wah, const bf16_t* __restrict__ Wob,
    const float* __restrict__ bo, bf16_t* __restrict__ hbuf,
    float* __restrict__ nst, float* __restrict__ dst,
    float* __restrict__ hst, float* __restrict__ amst,
    unsigned int* __restrict__ bar, float* __restrict__ out) {
  const int wg    = blockIdx.x;
  const int lane  = threadIdx.x & 31;
  const int wave  = threadIdx.x >> 5;
  const int cbase = wg * 64;

  __shared__ float lds_g[64 * 64];
  __shared__ float lds_a[64 * 64];

  for (int t = 0; t < T_; ++t) {
    const bf16_t* hcur  = hbuf + (size_t)(t & 1) * B_ * C_;
    bf16_t*       hnext = hbuf + (size_t)((t + 1) & 1) * B_ * C_;

    // ---- (1) h-GEMMs: [64b x 64c] for g (waves 0-3) and a (waves 4-7) -----
    {
      const int mat    = wave >> 2;
      const int mstrip = wave & 3;
      const bf16_t* Wm = mat ? Wah : Wgh;
      float* ldsout    = mat ? lds_a : lds_g;
      const int arow   = mstrip * 16 + (lane & 15);
      const int bcol   = cbase + (lane & 15);

      v8f acc[4] = {v8f{}, v8f{}, v8f{}, v8f{}};
      v16bf a_c = load_frag_b16(hcur, arow, C_, 0, lane);
      v16bf b_c[4];
#pragma unroll
      for (int j = 0; j < 4; ++j) b_c[j] = load_frag_b16(Wm, bcol + j * 16, C_, 0, lane);

      for (int k = 0; k < C_ - 32; k += 32) {          // pipelined main loop
        v16bf a_n = load_frag_b16(hcur, arow, C_, k + 32, lane);
        v16bf b_n[4];
#pragma unroll
        for (int j = 0; j < 4; ++j) b_n[j] = load_frag_b16(Wm, bcol + j * 16, C_, k + 32, lane);
#pragma unroll
        for (int j = 0; j < 4; ++j) acc[j] = wmma_bf16(a_c, b_c[j], acc[j]);
        a_c = a_n;
#pragma unroll
        for (int j = 0; j < 4; ++j) b_c[j] = b_n[j];
      }
#pragma unroll
      for (int j = 0; j < 4; ++j) acc[j] = wmma_bf16(a_c, b_c[j], acc[j]);

      const int m0 = mstrip * 16 + ((lane < 16) ? 0 : 8);
#pragma unroll
      for (int j = 0; j < 4; ++j) {
        int cl = j * 16 + (lane & 15);
#pragma unroll
        for (int r = 0; r < 8; ++r) ldsout[(m0 + r) * 64 + cl] = acc[j][r];
      }
    }
    __syncthreads();

    // ---- (2) elementwise scan update (own slice); Y is read-once -> NT ----
    for (int idx = threadIdx.x; idx < 64 * 64; idx += 256) {
      int b  = idx >> 6;
      int cl = idx & 63;
      int c  = cbase + cl;
      size_t yb = ((size_t)b * T_ + t) * NCAT_;
      float u  = __builtin_nontemporal_load(&Y[yb + 0 * C_ + c]);
      float g  = __builtin_nontemporal_load(&Y[yb + 1 * C_ + c]) + lds_g[b * 64 + cl];
      float av = __builtin_nontemporal_load(&Y[yb + 2 * C_ + c]) + lds_a[b * 64 + cl];
      float ed = __builtin_nontemporal_load(&Y[yb + 3 * C_ + c]);  // exp(tanh(decay))
      size_t si = (size_t)b * C_ + c;
      float nv = nst[si], dv = dst[si], am = amst[si];
      float z   = u * tanhf(g);
      float anm = fmaxf(am * ed, av);
      float e1  = __expf(am - anm);
      float e2  = __expf(av - anm);
      float nn  = nv * ed * e1 + z * e2;
      float dd  = dv * ed * e1 + e2;
      float hh  = tanhf(nn / dd);
      nst[si] = nn; dst[si] = dd; amst[si] = anm; hst[si] = hh;
      hnext[si] = (bf16_t)hh;
    }
    __syncthreads();

    // ---- (3) device-wide step barrier -------------------------------------
    if (threadIdx.x == 0) {
      __threadfence();                                 // release h_{t+1}
      atomicAdd(bar, 1u);
      unsigned target = (unsigned)(t + 1) * NWG_;
      while (atomicAdd(bar, 0u) < target) { }
      __threadfence();                                 // acquire
    }
    __syncthreads();
    __builtin_amdgcn_s_cluster_barrier();              // hw path when clustered

    // ---- (4) output GEMM: o_t slice [64(B) x 16(K)], pipelined ------------
    if (wave < 4) {
      const int arow = wave * 16 + (lane & 15);
      const int ncol = wg * 16 + (lane & 15);
      v8f acc = v8f{};
      v16bf a_c = load_frag_b16(hnext, arow, C_, 0, lane);
      v16bf b_c = load_frag_b16(Wob,   ncol, C_, 0, lane);
      for (int k = 0; k < C_ - 32; k += 32) {
        v16bf a_n = load_frag_b16(hnext, arow, C_, k + 32, lane);
        v16bf b_n = load_frag_b16(Wob,   ncol, C_, k + 32, lane);
        acc = wmma_bf16(a_c, b_c, acc);
        a_c = a_n; b_c = b_n;
      }
      acc = wmma_bf16(a_c, b_c, acc);

      const int m0 = wave * 16 + ((lane < 16) ? 0 : 8);
      float bov = bo[ncol];
#pragma unroll
      for (int r = 0; r < 8; ++r) {
        int b = m0 + r;
        __builtin_nontemporal_store(acc[r] + bov,
                                    &out[((size_t)b * T_ + t) * K_ + ncol]);
      }
    }
    __syncthreads();
  }
}

// Copy final recurrent state into the output tuple segments.
__global__ void finalize_kernel(const float* __restrict__ nst, const float* __restrict__ dst,
                                const float* __restrict__ hst, const float* __restrict__ amst,
                                float* __restrict__ on, float* __restrict__ od,
                                float* __restrict__ oh, float* __restrict__ oam) {
  int i = blockIdx.x * blockDim.x + threadIdx.x;
  if (i < B_ * C_) { on[i] = nst[i]; od[i] = dst[i]; oh[i] = hst[i]; oam[i] = amst[i]; }
}

// ---------------------------------------------------------------------------
extern "C" void kernel_launch(void* const* d_in, const int* in_sizes, int n_in,
                              void* d_out, int out_size, void* d_ws, size_t ws_size,
                              hipStream_t stream) {
  const float* x   = (const float*)d_in[0];
  const float* s   = (const float*)d_in[1];
  const float* n0  = (const float*)d_in[2];
  const float* d0  = (const float*)d_in[3];
  const float* h0  = (const float*)d_in[4];
  const float* am0 = (const float*)d_in[5];
  const float* Wg  = (const float*)d_in[6];
  const float* bg  = (const float*)d_in[7];
  const float* Wu  = (const float*)d_in[8];
  const float* bu  = (const float*)d_in[9];
  const float* Wa  = (const float*)d_in[10];
  const float* Wo  = (const float*)d_in[11];
  const float* bo  = (const float*)d_in[12];
  const float* Wd  = (const float*)d_in[13];
  float* out = (float*)d_out;

  char* base = (char*)d_ws;
  size_t off = 0;
  auto carve = [&](size_t bytes) -> void* {
    void* p = base + off;
    off += (bytes + 255) & ~(size_t)255;
    return p;
  };
  float*    Y    = (float*)carve((size_t)M_ * NCAT_ * sizeof(float));   // 512 MB
  bf16_t*   Xbf  = (bf16_t*)carve((size_t)M_ * F_ * 2);
  bf16_t*   Wcat = (bf16_t*)carve((size_t)NCAT_ * F_ * 2);
  bf16_t*   Wgh  = (bf16_t*)carve((size_t)C_ * C_ * 2);
  bf16_t*   Wah  = (bf16_t*)carve((size_t)C_ * C_ * 2);
  bf16_t*   Wob  = (bf16_t*)carve((size_t)K_ * C_ * 2);
  bf16_t*   hbuf = (bf16_t*)carve((size_t)2 * B_ * C_ * 2);             // ping-pong
  float*    nst  = (float*)carve((size_t)B_ * C_ * sizeof(float));
  float*    dst  = (float*)carve((size_t)B_ * C_ * sizeof(float));
  float*    hst  = (float*)carve((size_t)B_ * C_ * sizeof(float));
  float*    amst = (float*)carve((size_t)B_ * C_ * sizeof(float));
  unsigned* bar  = (unsigned*)carve(256);

  hipMemsetAsync(bar, 0, 256, stream);   // reset step barrier every call/replay

  pack_x_kernel<<<4096, 256, 0, stream>>>(x, Xbf);
  pack_w_kernel<<<4096, 256, 0, stream>>>(Wu, Wg, Wa, Wd, Wo, Wcat, Wgh, Wah, Wob);

  const size_t outs_elems = (size_t)B_ * T_ * K_;     // 8,388,608
  init_kernel<<<(B_ * C_ + 255) / 256, 256, 0, stream>>>(
      s, n0, d0, h0, am0, hbuf, nst, dst, hst, amst, out + outs_elems);

  dim3 g2(M_ / 256, NCAT_ / 64);                      // 128 x 64 blocks
  xgemm_kernel<<<g2, 256, 0, stream>>>(Xbf, Wcat, bu, bg, Y);

  recurrent_kernel<<<NWG_, 256, 0, stream>>>(Y, Wgh, Wah, Wob, bo, hbuf,
                                             nst, dst, hst, amst, bar, out);

  const size_t so = outs_elems + C_;                  // after `s`
  finalize_kernel<<<(B_ * C_ + 255) / 256, 256, 0, stream>>>(
      nst, dst, hst, amst,
      out + so, out + so + (size_t)B_ * C_,
      out + so + 2 * (size_t)B_ * C_, out + so + 3 * (size_t)B_ * C_);
}
